// MMD_loss_68375879352722
// MI455X (gfx1250) — matrix operand order repaired and checked by reading, not compiled, not run on hardware
//
#include <hip/hip_runtime.h>
#include <hip/hip_bf16.h>

// ---------------------------------------------------------------------------
// Gaussian MMD, fused WMMA implementation for gfx1250 (MI455X).
//   total = [source; target]  : 8192 x 256 f32
//   G = total @ total^T via v_wmma_f32_16x16x32_bf16 with hi/lo bf16 split
//   l2 = max(sq_i + sq_j - 2G, 0), 5-bandwidth exp sum fused in epilogue.
//   bandwidth computed in closed form: sum(l2) = 2n*sum(sq) - 2*||colsum||^2.
// ---------------------------------------------------------------------------

typedef __attribute__((ext_vector_type(16))) __bf16 v16bf;
typedef __attribute__((ext_vector_type(8)))  __bf16 v8bf;
typedef __attribute__((ext_vector_type(8)))  float  v8f;

#define MMD_N     8192
#define MMD_D     256
#define MMD_NSRC  4096

// Workspace layout (bytes)
#define OFF_HI      ((size_t)0)                         // 8192*256 bf16 = 4 MB
#define OFF_LO      ((size_t)4194304)                   // 4 MB
#define OFF_SQ      ((size_t)8388608)                   // 8192 f32
#define OFF_COLSUM  ((size_t)8421376)                   // 256 f32
#define OFF_SUMSQ   ((size_t)8422400)                   // double
#define OFF_ACC     ((size_t)8422408)                   // double
#define OFF_NC      ((size_t)8422416)                   // 8 f32 (5 used)
#define WS_END      ((size_t)8422448)

// ---------------------------------------------------------------------------
// Prep: split every f32 of `total` into hi(bf16) + lo(bf16), compute per-row
// squared norms, accumulate column sums and total sum-of-squares (for the
// closed-form bandwidth).
// ---------------------------------------------------------------------------
__global__ void __launch_bounds__(256)
mmd_prep(const float* __restrict__ src, const float* __restrict__ tgt,
         __bf16* __restrict__ hi, __bf16* __restrict__ lo,
         float* __restrict__ sq, float* __restrict__ colsum,
         double* __restrict__ sumsq) {
    const int r = blockIdx.x;          // row of `total`
    const int t = threadIdx.x;         // column, D == blockDim.x == 256
    const float v = (r < MMD_NSRC) ? src[r * MMD_D + t]
                                   : tgt[(r - MMD_NSRC) * MMD_D + t];
    const __bf16 h = (__bf16)v;
    hi[r * MMD_D + t] = h;
    lo[r * MMD_D + t] = (__bf16)(v - (float)h);
    atomicAdd(&colsum[t], v);

    __shared__ float red[256];
    red[t] = v * v;
    __syncthreads();
    for (int s = 128; s > 0; s >>= 1) {
        if (t < s) red[t] += red[t + s];
        __syncthreads();
    }
    if (t == 0) {
        sq[r] = red[0];
        atomicAdd(sumsq, (double)red[0]);
    }
}

// ---------------------------------------------------------------------------
// Bandwidth:  sum(l2) = 2*n*sum(sq) - 2*||colsum||^2   (clamp is a no-op in
// exact arithmetic since l2 >= 0).  Emit the 5 exp2 coefficients
// nc[t] = -log2(e) / (bandwidth * 2^t).
// ---------------------------------------------------------------------------
__global__ void __launch_bounds__(256)
mmd_bandwidth(const float* __restrict__ colsum, const double* __restrict__ sumsq,
              float* __restrict__ nc) {
    __shared__ float red[256];
    const int t = threadIdx.x;
    const float c = colsum[t];
    red[t] = c * c;
    __syncthreads();
    for (int s = 128; s > 0; s >>= 1) {
        if (t < s) red[t] += red[t + s];
        __syncthreads();
    }
    if (t == 0) {
        const double n  = (double)MMD_N;
        const double l2sum = 2.0 * n * (*sumsq) - 2.0 * (double)red[0];
        double bw = l2sum / (n * n - n);
        bw *= 0.25;                       // / KERNEL_MUL^(KERNEL_NUM//2) = 2^2
        const double LOG2E = 1.4426950408889634;
        for (int i = 0; i < 5; ++i)
            nc[i] = (float)(-LOG2E / (bw * (double)(1 << i)));
    }
}

// ---------------------------------------------------------------------------
// Main fused kernel.  Each wave owns one 16x16 output tile (ti, tj), tj >= ti.
// Block = 256 threads = 8 waves covering a 64x32 tile region.
//   A operand = rows [ti*16, ti*16+16) of T   (16x32 bf16, ISA A layout)
//   B operand = rows [tj*16, tj*16+16) of T   (= T^T columns; 32x16 layout,
//               each lane reads two runs of 8 contiguous bf16 -> coalesced)
// hi/lo split: G ~= hi*hi + hi*lo + lo*hi  (3 WMMAs per K=32 step, 24/tile).
// ---------------------------------------------------------------------------
__global__ void __launch_bounds__(256)
mmd_tiles(const __bf16* __restrict__ hi, const __bf16* __restrict__ lo,
          const float* __restrict__ sq, const float* __restrict__ nc,
          double* __restrict__ acc) {
    const int lane = threadIdx.x & 31;
    const int wave = threadIdx.x >> 5;                 // 0..7
    const int ti = blockIdx.y * 4 + (wave >> 1);       // 0..511
    const int tj = blockIdx.x * 2 + (wave & 1);        // 0..511
    if (tj < ti) return;                               // symmetric: upper only

    const int half = lane >> 4;                        // 0 | 1
    const int rA = ti * 16 + (lane & 15);
    const int rB = tj * 16 + (lane & 15);
    const __bf16* __restrict__ hiA = hi + (size_t)rA * MMD_D;
    const __bf16* __restrict__ loA = lo + (size_t)rA * MMD_D;
    const __bf16* __restrict__ hiB = hi + (size_t)rB * MMD_D;
    const __bf16* __restrict__ loB = lo + (size_t)rB * MMD_D;

    v8f c = {};
#pragma unroll
    for (int kc = 0; kc < MMD_D / 32; ++kc) {
        const int k0 = kc * 32;
        // A 16x32 bf16 layout: lane L, elems 0..7 -> K = half*8 + e,
        //                               elems 8..15 -> K = 16 + half*8 + (e-8)
        const v8bf ah0 = *(const v8bf*)(hiA + k0 + half * 8);
        const v8bf ah1 = *(const v8bf*)(hiA + k0 + 16 + half * 8);
        const v8bf al0 = *(const v8bf*)(loA + k0 + half * 8);
        const v8bf al1 = *(const v8bf*)(loA + k0 + 16 + half * 8);
        const v16bf a_hi = __builtin_shufflevector(ah0, ah1,
            0,1,2,3,4,5,6,7,8,9,10,11,12,13,14,15);
        const v16bf a_lo = __builtin_shufflevector(al0, al1,
            0,1,2,3,4,5,6,7,8,9,10,11,12,13,14,15);
        // B 32x16 bf16 layout: lane L holds column n = L&15,
        // same K order as A: elems 0..7 -> K = half*8 + e, elems 8..15 -> K = 16 + half*8 + (e-8).
        const v8bf bh0 = *(const v8bf*)(hiB + k0 + half * 8);
        const v8bf bh1 = *(const v8bf*)(hiB + k0 + 16 + half * 8);
        const v8bf bl0 = *(const v8bf*)(loB + k0 + half * 8);
        const v8bf bl1 = *(const v8bf*)(loB + k0 + 16 + half * 8);
        const v16bf b_hi = __builtin_shufflevector(bh0, bh1,
            0,1,2,3,4,5,6,7,8,9,10,11,12,13,14,15);
        const v16bf b_lo = __builtin_shufflevector(bl0, bl1,
            0,1,2,3,4,5,6,7,8,9,10,11,12,13,14,15);

        c = __builtin_amdgcn_wmma_f32_16x16x32_bf16(false, a_hi, false, b_hi,
                                                    (short)0, c, false, false);
        c = __builtin_amdgcn_wmma_f32_16x16x32_bf16(false, a_hi, false, b_lo,
                                                    (short)0, c, false, false);
        c = __builtin_amdgcn_wmma_f32_16x16x32_bf16(false, a_lo, false, b_hi,
                                                    (short)0, c, false, false);
    }

    // Epilogue: l2 -> 5-bandwidth exp sum (1 exp2 each; -O3 lowers to v_exp_f32)
    const float nc0 = nc[0], nc1 = nc[1], nc2 = nc[2], nc3 = nc[3], nc4 = nc[4];
    const float sqj = sq[tj * 16 + (lane & 15)];
    float s = 0.0f;
#pragma unroll
    for (int r = 0; r < 8; ++r) {
        const int m = r + half * 8;                    // C/D: M = r + 8*half
        const float sqi = sq[ti * 16 + m];
        const float l2 = fmaxf(sqi + sqj - 2.0f * c[r], 0.0f);
        s += exp2f(l2 * nc0) + exp2f(l2 * nc1) + exp2f(l2 * nc2)
           + exp2f(l2 * nc3) + exp2f(l2 * nc4);
    }
    // wave32 butterfly reduction
#pragma unroll
    for (int off = 16; off > 0; off >>= 1)
        s += __shfl_xor(s, off, 32);

    if (lane == 0) {
        const float sign = ((ti < MMD_NSRC / 16) == (tj < MMD_NSRC / 16))
                               ? 1.0f : -1.0f;         // xx,yy: +   xy,yx: -
        const float w = (ti == tj) ? 1.0f : 2.0f;      // symmetry weight
        atomicAdd(acc, (double)(sign * w) * (double)s);
    }
}

__global__ void mmd_finalize(const double* __restrict__ acc, float* __restrict__ out) {
    if (threadIdx.x == 0 && blockIdx.x == 0)
        out[0] = (float)(acc[0] * (1.0 / ((double)MMD_NSRC * (double)MMD_NSRC)));
}

// ---------------------------------------------------------------------------
extern "C" void kernel_launch(void* const* d_in, const int* in_sizes, int n_in,
                              void* d_out, int out_size, void* d_ws, size_t ws_size,
                              hipStream_t stream) {
    const float* src = (const float*)d_in[0];
    const float* tgt = (const float*)d_in[1];
    float* out = (float*)d_out;
    char* ws = (char*)d_ws;

    __bf16* hi     = (__bf16*)(ws + OFF_HI);
    __bf16* lo     = (__bf16*)(ws + OFF_LO);
    float*  sq     = (float*)(ws + OFF_SQ);
    float*  colsum = (float*)(ws + OFF_COLSUM);
    double* sumsq  = (double*)(ws + OFF_SUMSQ);
    double* acc    = (double*)(ws + OFF_ACC);
    float*  nc     = (float*)(ws + OFF_NC);

    // Zero the small accumulator region every call (graph-capture safe).
    hipMemsetAsync(ws + OFF_COLSUM, 0, WS_END - OFF_COLSUM, stream);

    mmd_prep<<<MMD_N, 256, 0, stream>>>(src, tgt, hi, lo, sq, colsum, sumsq);
    mmd_bandwidth<<<1, 256, 0, stream>>>(colsum, sumsq, nc);

    dim3 grid(MMD_N / 32, MMD_N / 64);   // (tj blocks of 2 tiles, ti blocks of 4)
    mmd_tiles<<<grid, 256, 0, stream>>>(hi, lo, sq, nc, acc);

    mmd_finalize<<<1, 1, 0, stream>>>(acc, out);
}
